// SimpleAttention_37881611550827
// MI455X (gfx1250) — matrix operand order repaired and checked by reading, not compiled, not run on hardware
//
#include <hip/hip_runtime.h>
#include <hip/hip_bf16.h>

typedef __bf16 bf16_t;
typedef __attribute__((ext_vector_type(16))) __bf16 v16bf;
typedef __attribute__((ext_vector_type(8)))  __bf16 v8bf;
typedef __attribute__((ext_vector_type(8)))  float   v8f;

#define B_ 8
#define N_ 1024
#define C_ 1024
#define H_ 16
#define D_ 64
#define SCALE_ 0.125f   // 1/sqrt(64)

#define WMMA_BF16(a, b, c) \
  __builtin_amdgcn_wmma_f32_16x16x32_bf16(false, (a), false, (b), (short)0, (c), false, false)

// ---------------------------------------------------------------------------
// gfx1250 async global->LDS copy (ASYNCcnt path), with portable fallback.
// ROCm 7.2 clang-22 signature: (int4 AS(1)* src, int4 AS(3)* dst, imm, imm).
// ---------------------------------------------------------------------------
#if defined(__gfx1250__) && __has_builtin(__builtin_amdgcn_global_load_async_to_lds_b128)
#define HAVE_ASYNC_LDS 1
typedef int v4i_vs __attribute__((vector_size(16)));
typedef __attribute__((address_space(1))) v4i_vs* gbl_v4i_p;
typedef __attribute__((address_space(3))) v4i_vs* lds_v4i_p;
#else
#define HAVE_ASYNC_LDS 0
#endif

__device__ __forceinline__ void async_copy16(const bf16_t* g, bf16_t* l) {
#if HAVE_ASYNC_LDS
  __builtin_amdgcn_global_load_async_to_lds_b128((gbl_v4i_p)g, (lds_v4i_p)l, 0, 0);
#else
  *(v8bf*)l = *(const v8bf*)g;
#endif
}

template <int Nkeep>
__device__ __forceinline__ void async_wait() {
#if HAVE_ASYNC_LDS
#if __has_builtin(__builtin_amdgcn_s_wait_asynccnt)
  __builtin_amdgcn_s_wait_asynccnt(Nkeep);
#else
  if (Nkeep == 0) asm volatile("s_wait_asynccnt 0x0" ::: "memory");
  else            asm volatile("s_wait_asynccnt 0x1" ::: "memory");
#endif
#endif
}

// ---------------------------------------------------------------------------
// Fragment loaders for v_wmma_f32_16x16x32_bf16 (wave32).
// A (16x32, MxK): lanes 0-15 row M=lane, K runs {0..7,16..23}; lanes 16-31
//                 same rows, K runs {8..15,24..31}.  (ISA 7.12.2)
// B (32x16, KxN): lane&15 = N col; lanes 0-15 hold K {0..7,16..23},
//                 lanes 16-31 hold K {8..15,24..31}.  (matches A)
// Both assume K contiguous in memory (stride = row pitch in elements).
// ---------------------------------------------------------------------------
__device__ __forceinline__ v16bf load_a_k32(const bf16_t* __restrict__ base, int stride) {
  const int lane = threadIdx.x & 31;
  const int r = lane & 15, hv = lane >> 4;
  const bf16_t* p = base + (size_t)r * stride + hv * 8;
  v8bf lo = *(const v8bf*)(p);
  v8bf hi = *(const v8bf*)(p + 16);
  v16bf out;
#pragma unroll
  for (int i = 0; i < 8; ++i) { out[i] = lo[i]; out[8 + i] = hi[i]; }
  return out;
}

__device__ __forceinline__ v16bf load_b_k32(const bf16_t* __restrict__ base, int stride) {
  const int lane = threadIdx.x & 31;
  const int n = lane & 15, hv = lane >> 4;
  const bf16_t* p = base + (size_t)n * stride + hv * 8;
  v8bf lo = *(const v8bf*)(p);
  v8bf hi = *(const v8bf*)(p + 16);
  v16bf out;
#pragma unroll
  for (int i = 0; i < 8; ++i) { out[i] = lo[i]; out[8 + i] = hi[i]; }
  return out;
}

// ---------------------------------------------------------------------------
// fp32 -> bf16 conversion
// ---------------------------------------------------------------------------
__global__ __launch_bounds__(256) void cvt_f32_bf16(const float* __restrict__ in,
                                                    bf16_t* __restrict__ out, int n) {
  int i = blockIdx.x * blockDim.x + threadIdx.x;
  if (i < n) out[i] = (bf16_t)in[i];
}

// ---------------------------------------------------------------------------
// GEMM core: acc[2][4] (32 rows x 64 cols per wave), W tile staged in LDS
// (double-buffered, async), A fragments register-prefetched one k-step ahead.
// Block = 256 threads = 8 waves -> 256x64 block tile.
// ---------------------------------------------------------------------------
__device__ __forceinline__ void gemm_core(const bf16_t* __restrict__ A,
                                          const bf16_t* __restrict__ W,
                                          int K, int row0, int col0,
                                          bf16_t (*Wlds)[64][32],   // [2][64][32]
                                          v8f acc[2][4]) {
  // W staging role: 256 threads cover 64 cols x 4 chunks of 8 elems (16B each)
  const int scol = threadIdx.x >> 2;        // 0..63
  const int sk   = (threadIdx.x & 3) * 8;   // 0,8,16,24
  const bf16_t* wsrc = W + (size_t)(col0 + scol) * K + sk;

  const bf16_t* a0 = A + (size_t)row0 * K;
  const bf16_t* a1 = A + (size_t)(row0 + 16) * K;

  const int nsteps = K / 32;

  // stage k-step 0 into buffer 0; preload A frags for k-step 0
  async_copy16(wsrc, &Wlds[0][scol][sk]);
  v16bf afc0 = load_a_k32(a0, K);
  v16bf afc1 = load_a_k32(a1, K);

  for (int i = 0; i < nsteps; ++i) {
    const int buf = i & 1;
    if (i + 1 < nsteps) {
      async_copy16(wsrc + (i + 1) * 32, &Wlds[buf ^ 1][scol][sk]);
      async_wait<1>();              // batch for buf is complete
    } else {
      async_wait<0>();
    }
    __syncthreads();                // W tile (buf) visible to all waves

    // register-prefetch next A fragments (overlaps with WMMAs below)
    const int kn = (i + 1 < nsteps) ? (i + 1) * 32 : 0;
    v16bf afn0 = load_a_k32(a0 + kn, K);
    v16bf afn1 = load_a_k32(a1 + kn, K);

    v16bf bf0 = load_b_k32(&Wlds[buf][0][0],  32);
    v16bf bf1 = load_b_k32(&Wlds[buf][16][0], 32);
    v16bf bf2 = load_b_k32(&Wlds[buf][32][0], 32);
    v16bf bf3 = load_b_k32(&Wlds[buf][48][0], 32);

    acc[0][0] = WMMA_BF16(afc0, bf0, acc[0][0]);
    acc[0][1] = WMMA_BF16(afc0, bf1, acc[0][1]);
    acc[0][2] = WMMA_BF16(afc0, bf2, acc[0][2]);
    acc[0][3] = WMMA_BF16(afc0, bf3, acc[0][3]);
    acc[1][0] = WMMA_BF16(afc1, bf0, acc[1][0]);
    acc[1][1] = WMMA_BF16(afc1, bf1, acc[1][1]);
    acc[1][2] = WMMA_BF16(afc1, bf2, acc[1][2]);
    acc[1][3] = WMMA_BF16(afc1, bf3, acc[1][3]);

    afc0 = afn0; afc1 = afn1;
    __syncthreads();                // all reads of buf done before it is restaged
  }
}

// Out[M,Nout](bf16) = A[M,K](bf16) @ W[Nout,K]^T(bf16)
__global__ __launch_bounds__(256) void gemm_bf16out(const bf16_t* __restrict__ A,
                                                    const bf16_t* __restrict__ W,
                                                    bf16_t* __restrict__ Out,
                                                    int M, int Nout, int K) {
  __shared__ __align__(16) bf16_t Wlds[2][64][32];
  const int wave = threadIdx.x >> 5;
  const int lane = threadIdx.x & 31;
  const int r15 = lane & 15, hv = lane >> 4;
  const int row0 = blockIdx.y * 256 + wave * 32;
  const int col0 = blockIdx.x * 64;

  v8f acc[2][4] = {};
  gemm_core(A, W, K, row0, col0, Wlds, acc);

#pragma unroll
  for (int g = 0; g < 2; ++g)
#pragma unroll
    for (int t = 0; t < 4; ++t)
#pragma unroll
      for (int r = 0; r < 8; ++r)
        Out[(size_t)(row0 + g * 16 + r + hv * 8) * Nout + col0 + t * 16 + r15] =
            (bf16_t)acc[g][t][r];
}

// Same GEMM, fp32 output with bias (final projection)
__global__ __launch_bounds__(256) void gemm_f32out(const bf16_t* __restrict__ A,
                                                   const bf16_t* __restrict__ W,
                                                   const float* __restrict__ bias,
                                                   float* __restrict__ Out,
                                                   int M, int Nout, int K) {
  __shared__ __align__(16) bf16_t Wlds[2][64][32];
  const int wave = threadIdx.x >> 5;
  const int lane = threadIdx.x & 31;
  const int r15 = lane & 15, hv = lane >> 4;
  const int row0 = blockIdx.y * 256 + wave * 32;
  const int col0 = blockIdx.x * 64;

  v8f acc[2][4] = {};
  gemm_core(A, W, K, row0, col0, Wlds, acc);

#pragma unroll
  for (int t = 0; t < 4; ++t) {
    float bv = bias[col0 + t * 16 + r15];
#pragma unroll
    for (int g = 0; g < 2; ++g)
#pragma unroll
      for (int r = 0; r < 8; ++r)
        Out[(size_t)(row0 + g * 16 + r + hv * 8) * Nout + col0 + t * 16 + r15] =
            acc[g][t][r] + bv;
  }
}

// ---------------------------------------------------------------------------
// Flash attention: grid(N/64, B*H), block 128 (4 waves, 16 q-rows each).
// qkv layout: [B*N, 3C]; Q at col h*64, K at C + h*64, V at 2C + h*64.
// K-fragments and V-registers are pipelined one 32-key block ahead.
// ---------------------------------------------------------------------------
__global__ __launch_bounds__(128) void attn_kernel(const bf16_t* __restrict__ qkv,
                                                   bf16_t* __restrict__ attnout) {
  __shared__ __align__(16) bf16_t Plds[4][16][32];  // per-wave P tile (16q x 32k)
  __shared__ __align__(16) bf16_t Vt[64][32];       // V block transposed: [d][key]

  const int wave = threadIdx.x >> 5;
  const int lane = threadIdx.x & 31;
  const int r15 = lane & 15, hv = lane >> 4;
  const int bh = blockIdx.y;
  const int b = bh / H_, h = bh % H_;
  const int q0 = blockIdx.x * 64 + wave * 16;

  const int rowstride = 3 * C_;
  const bf16_t* qbase  = qkv + (size_t)(b * N_ + q0) * rowstride + h * D_;
  const bf16_t* kbase0 = qkv + (size_t)(b * N_) * rowstride + C_ + h * D_;
  const bf16_t* vbase0 = qkv + (size_t)(b * N_) * rowstride + 2 * C_ + h * D_;

  // Q fragments (d = 0..31 / 32..63), reused across all key blocks
  v16bf qf0 = load_a_k32(qbase + 0, rowstride);
  v16bf qf1 = load_a_k32(qbase + 32, rowstride);

  v8f o0 = {}, o1 = {}, o2 = {}, o3 = {};
  float m[8], l[8];
#pragma unroll
  for (int r = 0; r < 8; ++r) { m[r] = -3.0e38f; l[r] = 0.0f; }

  // cooperative V staging role: 128 threads cover 32 keys x 4 d-chunks of 16
  const int skey = threadIdx.x >> 2;
  const int sd   = (threadIdx.x & 3) * 16;

  // ---- pipelined preloads for key block 0 ----
  v16bf kf0, kf1, kf2, kf3;
  {
    kf0 = load_b_k32(kbase0, rowstride);
    kf1 = load_b_k32(kbase0 + 32, rowstride);
    kf2 = load_b_k32(kbase0 + (size_t)16 * rowstride, rowstride);
    kf3 = load_b_k32(kbase0 + (size_t)16 * rowstride + 32, rowstride);
  }
  v8bf va, vb;
  {
    const bf16_t* vp = vbase0 + (size_t)skey * rowstride + sd;
    va = *(const v8bf*)(vp);
    vb = *(const v8bf*)(vp + 8);
  }

  for (int kb = 0; kb < N_; kb += 32) {
    // ---- stage V block (already in registers) transposed into LDS ----
#pragma unroll
    for (int j = 0; j < 8; ++j) {
      Vt[sd + j][skey]     = va[j];
      Vt[sd + 8 + j][skey] = vb[j];
    }

    // ---- issue next block's K fragments + V registers (overlap compute) ----
    const int kbn = (kb + 32 < N_) ? kb + 32 : 0;
    v16bf kn0, kn1, kn2, kn3;
    {
      const bf16_t* kbp = kbase0 + (size_t)kbn * rowstride;
      kn0 = load_b_k32(kbp, rowstride);
      kn1 = load_b_k32(kbp + 32, rowstride);
      kn2 = load_b_k32(kbp + (size_t)16 * rowstride, rowstride);
      kn3 = load_b_k32(kbp + (size_t)16 * rowstride + 32, rowstride);
    }
    v8bf van, vbn;
    {
      const bf16_t* vp = vbase0 + (size_t)(kbn + skey) * rowstride + sd;
      van = *(const v8bf*)(vp);
      vbn = *(const v8bf*)(vp + 8);
    }

    // ---- S = Q K^T (two 16-key column tiles) ----
    v8f s0 = {}, s1 = {};
    s0 = WMMA_BF16(qf0, kf0, s0);
    s0 = WMMA_BF16(qf1, kf1, s0);
    s1 = WMMA_BF16(qf0, kf2, s1);
    s1 = WMMA_BF16(qf1, kf3, s1);

    // ---- online softmax (row stats in 16-lane groups) ----
#pragma unroll
    for (int r = 0; r < 8; ++r) {
      float a0 = s0[r] * SCALE_;
      float a1 = s1[r] * SCALE_;
      float rm = fmaxf(a0, a1);
      rm = fmaxf(rm, __shfl_xor(rm, 1));
      rm = fmaxf(rm, __shfl_xor(rm, 2));
      rm = fmaxf(rm, __shfl_xor(rm, 4));
      rm = fmaxf(rm, __shfl_xor(rm, 8));
      float mnew  = fmaxf(m[r], rm);
      float alpha = __expf(m[r] - mnew);
      m[r] = mnew;
      float p0 = __expf(a0 - mnew);
      float p1 = __expf(a1 - mnew);
      float rs = p0 + p1;
      rs += __shfl_xor(rs, 1);
      rs += __shfl_xor(rs, 2);
      rs += __shfl_xor(rs, 4);
      rs += __shfl_xor(rs, 8);
      l[r] = l[r] * alpha + rs;
      o0[r] *= alpha; o1[r] *= alpha; o2[r] *= alpha; o3[r] *= alpha;
      // store P (C-layout -> LDS) for relayout into an A fragment
      Plds[wave][r + hv * 8][r15]      = (bf16_t)p0;
      Plds[wave][r + hv * 8][16 + r15] = (bf16_t)p1;
    }

    __syncthreads();  // Vt staged + Plds written

    // ---- O += P @ V ----
    v16bf pf  = load_a_k32(&Plds[wave][0][0], 32);
    v16bf vf0 = load_b_k32(&Vt[0][0],  32);
    v16bf vf1 = load_b_k32(&Vt[16][0], 32);
    v16bf vf2 = load_b_k32(&Vt[32][0], 32);
    v16bf vf3 = load_b_k32(&Vt[48][0], 32);
    o0 = WMMA_BF16(pf, vf0, o0);
    o1 = WMMA_BF16(pf, vf1, o1);
    o2 = WMMA_BF16(pf, vf2, o2);
    o3 = WMMA_BF16(pf, vf3, o3);

    __syncthreads();  // all LDS reads done before next iteration restages

    kf0 = kn0; kf1 = kn1; kf2 = kn2; kf3 = kn3;
    va = van; vb = vbn;
  }

  // ---- normalize and store [B*N, C] bf16 ----
#pragma unroll
  for (int r = 0; r < 8; ++r) {
    const size_t row = (size_t)(b * N_ + q0 + r + hv * 8);
    bf16_t* op = attnout + row * C_ + h * D_ + r15;
    float inv = 1.0f / l[r];
    op[0]  = (bf16_t)(o0[r] * inv);
    op[16] = (bf16_t)(o1[r] * inv);
    op[32] = (bf16_t)(o2[r] * inv);
    op[48] = (bf16_t)(o3[r] * inv);
  }
}

// ---------------------------------------------------------------------------
// Launcher
// ---------------------------------------------------------------------------
extern "C" void kernel_launch(void* const* d_in, const int* in_sizes, int n_in,
                              void* d_out, int out_size, void* d_ws, size_t ws_size,
                              hipStream_t stream) {
  (void)in_sizes; (void)n_in; (void)out_size; (void)ws_size;
  const float* x      = (const float*)d_in[0];
  const float* qkv_w  = (const float*)d_in[1];
  const float* proj_w = (const float*)d_in[2];
  const float* proj_b = (const float*)d_in[3];
  float* out = (float*)d_out;

  const int M = B_ * N_;
  char* ws = (char*)d_ws;
  const size_t sz_xb  = (size_t)M * C_ * 2;            // 16 MB
  const size_t sz_qw  = (size_t)3 * C_ * C_ * 2;       // 6 MB
  const size_t sz_pw  = (size_t)C_ * C_ * 2;           // 2 MB
  const size_t sz_qkv = (size_t)M * 3 * C_ * 2;        // 48 MB

  bf16_t* xb   = (bf16_t*)(ws);
  bf16_t* qwb  = (bf16_t*)(ws + sz_xb);
  bf16_t* pwb  = (bf16_t*)(ws + sz_xb + sz_qw);
  bf16_t* qkv  = (bf16_t*)(ws + sz_xb + sz_qw + sz_pw);
  bf16_t* aout = (bf16_t*)(ws + sz_xb + sz_qw + sz_pw + sz_qkv);

  // 1) convert fp32 -> bf16
  int nx = M * C_;
  cvt_f32_bf16<<<(nx + 255) / 256, 256, 0, stream>>>(x, xb, nx);
  int nq = 3 * C_ * C_;
  cvt_f32_bf16<<<(nq + 255) / 256, 256, 0, stream>>>(qkv_w, qwb, nq);
  int np = C_ * C_;
  cvt_f32_bf16<<<(np + 255) / 256, 256, 0, stream>>>(proj_w, pwb, np);

  // 2) QKV projection: [M,3C] = x @ qkv_w^T
  dim3 g1(3 * C_ / 64, M / 256);
  gemm_bf16out<<<g1, 256, 0, stream>>>(xb, qwb, qkv, M, 3 * C_, C_);

  // 3) flash attention
  dim3 g2(N_ / 64, B_ * H_);
  attn_kernel<<<g2, 128, 0, stream>>>(qkv, aout);

  // 4) output projection with bias (fp32 out)
  dim3 g3(C_ / 64, M / 256);
  gemm_f32out<<<g3, 256, 0, stream>>>(aout, pwb, proj_b, out, M, C_, C_);
}
